// TransformerBlock_21878563406222
// MI455X (gfx1250) — matrix operand order repaired and checked by reading, not compiled, not run on hardware
//
#include <hip/hip_runtime.h>
#include <cstdint>
#include <cstddef>

// ---------------------------------------------------------------------------
// Transformer block for MI455X (gfx1250, wave32, WMMA).
// All heavy math runs on v_wmma_f32_16x16x32_bf16 (bf16 in, f32 accumulate).
// ---------------------------------------------------------------------------

#define EMB    1024
#define SEQ    2048
#define NHEAD  16
#define HDIM   64
#define FFDIM  4096
#define ROWS   (2 * SEQ)   // 4096 total token rows (batch * seq)

typedef __bf16 bf16x16 __attribute__((ext_vector_type(16)));
typedef float  f32x8   __attribute__((ext_vector_type(8)));

union FragAB {
    bf16x16      v;
    unsigned int u[8];
};

static __device__ __forceinline__ unsigned short f2bf(float f) {
    // round-to-nearest-even f32 -> bf16
    unsigned int u = __float_as_uint(f);
    u += 0x7FFFu + ((u >> 16) & 1u);
    return (unsigned short)(u >> 16);
}

static __device__ __forceinline__ f32x8 wmma_bf16(const FragAB& a, const FragAB& b, f32x8 c) {
    // D = A(16x32 bf16) * B(32x16 bf16) + C(16x16 f32)
    return __builtin_amdgcn_wmma_f32_16x16x32_bf16(
        /*neg_a=*/false, a.v, /*neg_b=*/false, b.v,
        /*c_mod=*/(short)0, c, /*reuse_a=*/false, /*reuse_b=*/false);
}

// pack two bf16 rows into one dword: lo16 = rowEven, hi16 = rowOdd (per column)
#define PERM_EVEN 0x05040100u   // byte0,1 <- S1 bytes 0,1 ; byte2,3 <- S0 bytes 0,1
#define PERM_ODD  0x07060302u   // byte0,1 <- S1 bytes 2,3 ; byte2,3 <- S0 bytes 2,3

// ---------------------------------------------------------------------------
// f32 -> bf16 bulk conversion (4 elems / thread)
// ---------------------------------------------------------------------------
__global__ __launch_bounds__(256) void convert_bf16_kernel(
    const float* __restrict__ src, unsigned short* __restrict__ dst, int n4)
{
    int i = blockIdx.x * 256 + threadIdx.x;
    if (i < n4) {
        float4 v = reinterpret_cast<const float4*>(src)[i];
        uint2 r;
        r.x = (unsigned)f2bf(v.x) | ((unsigned)f2bf(v.y) << 16);
        r.y = (unsigned)f2bf(v.z) | ((unsigned)f2bf(v.w) << 16);
        reinterpret_cast<uint2*>(dst)[i] = r;
    }
}

// ---------------------------------------------------------------------------
// LayerNorm over 1024 cols, one row per 256-thread block, bf16 output
// ---------------------------------------------------------------------------
__global__ __launch_bounds__(256) void layernorm_bf16_kernel(
    const float* __restrict__ x, const float* __restrict__ sc,
    const float* __restrict__ sh, unsigned short* __restrict__ out)
{
    __shared__ float redS[256];
    __shared__ float redQ[256];
    const int row = blockIdx.x;
    const int tid = threadIdx.x;
    const float* xr = x + (size_t)row * EMB;

    float v[4];
    float s = 0.0f, q = 0.0f;
#pragma unroll
    for (int i = 0; i < 4; ++i) {
        v[i] = xr[tid + i * 256];
        s += v[i];
        q += v[i] * v[i];
    }
    redS[tid] = s; redQ[tid] = q;
    __syncthreads();
    for (int off = 128; off > 0; off >>= 1) {
        if (tid < off) { redS[tid] += redS[tid + off]; redQ[tid] += redQ[tid + off]; }
        __syncthreads();
    }
    const float mean = redS[0] * (1.0f / EMB);
    const float var  = redQ[0] * (1.0f / EMB) - mean * mean;
    const float rstd = rsqrtf(var + 1e-5f);
    unsigned short* orow = out + (size_t)row * EMB;
#pragma unroll
    for (int i = 0; i < 4; ++i) {
        const int c = tid + i * 256;
        orow[c] = f2bf(sc[c] * ((v[i] - mean) * rstd) + sh[c]);
    }
}

// ---------------------------------------------------------------------------
// Tiled bf16 WMMA GEMM:  C[M,N] = A[M,K] * B[K,N]  (+bias, +GELU, +residual)
// Block tile 128x128, K-step 32, 8 waves (2x4), 8 WMMA tiles per wave.
// B tile is staged into LDS pre-packed as row-pairs (v_perm_b32) so
// B-fragments are 8 aligned ds_load_b32 each.
// M%128==0, N%128==0, K%32==0 guaranteed by launch shapes.
// ---------------------------------------------------------------------------
template<bool OUT_BF16, bool BIAS, bool GELU_ACT, bool RESID>
__global__ __launch_bounds__(256) void gemm_bf16_kernel(
    const unsigned short* __restrict__ A,
    const unsigned short* __restrict__ B,
    const float* __restrict__ bias,
    const float* __restrict__ resid,
    void* __restrict__ outp,
    int M, int N, int K)
{
    constexpr int BM = 128, BN = 128, BK = 32;
    constexpr int LDA  = BK + 8;    // ushorts (even, padded vs bank conflicts)
    constexpr int LDBP = BN + 4;    // dwords per packed row-pair
    __shared__ unsigned short sA[BM * LDA];
    __shared__ unsigned int   sBp[(BK / 2) * LDBP];

    const int tid  = threadIdx.x;
    const int lane = tid & 31;
    const int wv   = tid >> 5;        // 0..7
    const int wm   = wv >> 2;         // 0..1 : 64-row strip
    const int wn   = wv & 3;          // 0..3 : 32-col strip
    const int half = lane >> 4;
    const int lm   = lane & 15;

    const int row0 = blockIdx.y * BM;
    const int col0 = blockIdx.x * BN;

    f32x8 acc[4][2];
#pragma unroll
    for (int i = 0; i < 4; ++i)
#pragma unroll
        for (int j = 0; j < 2; ++j)
#pragma unroll
            for (int r = 0; r < 8; ++r)
                acc[i][j][r] = 0.0f;

    const int ar = tid >> 1;          // 0..127   A stage: row
    const int ac = (tid & 1) * 16;    // 0/16     A stage: col
    const int bp = tid >> 4;          // 0..15    B stage: row pair
    const int bc = (tid & 15) * 8;    // 0..120   B stage: col

    for (int k0 = 0; k0 < K; k0 += BK) {
        // --- stage A tile (raw bf16 rows) ---
        {
            const uint4* g = reinterpret_cast<const uint4*>(A + (size_t)(row0 + ar) * K + k0 + ac);
            uint4 v0 = g[0], v1 = g[1];
            uint4* s = reinterpret_cast<uint4*>(&sA[ar * LDA + ac]);
            s[0] = v0; s[1] = v1;
        }
        // --- stage B tile packed as row-pairs: dword = rowEven | rowOdd<<16 ---
        {
            const size_t base = (size_t)(k0 + 2 * bp) * N + col0 + bc;
            const uint4* g0 = reinterpret_cast<const uint4*>(B + base);
            const uint4* g1 = reinterpret_cast<const uint4*>(B + base + N);
            uint4 r0 = g0[0], r1 = g1[0];
            uint4 o0, o1;
            o0.x = __builtin_amdgcn_perm(r1.x, r0.x, PERM_EVEN);
            o0.y = __builtin_amdgcn_perm(r1.x, r0.x, PERM_ODD);
            o0.z = __builtin_amdgcn_perm(r1.y, r0.y, PERM_EVEN);
            o0.w = __builtin_amdgcn_perm(r1.y, r0.y, PERM_ODD);
            o1.x = __builtin_amdgcn_perm(r1.z, r0.z, PERM_EVEN);
            o1.y = __builtin_amdgcn_perm(r1.z, r0.z, PERM_ODD);
            o1.z = __builtin_amdgcn_perm(r1.w, r0.w, PERM_EVEN);
            o1.w = __builtin_amdgcn_perm(r1.w, r0.w, PERM_ODD);
            uint4* s = reinterpret_cast<uint4*>(&sBp[bp * LDBP + bc]);
            s[0] = o0; s[1] = o1;
        }
        // prefetch next k-step tiles into L2 while we compute
        if (k0 + BK < K) {
            __builtin_prefetch(A + (size_t)(row0 + ar) * K + k0 + BK + ac, 0, 1);
            __builtin_prefetch(B + (size_t)(k0 + BK + 2 * bp) * N + col0 + bc, 0, 1);
        }
        __syncthreads();

        // --- B fragments: 8 x ds_load_b32 each (pair p = K/2) ---
        FragAB bfr[2];
#pragma unroll
        for (int tn = 0; tn < 2; ++tn) {
            const int n = wn * 32 + tn * 16 + lm;
#pragma unroll
            for (int i = 0; i < 8; ++i)
                bfr[tn].u[i] = sBp[((i & 3) + 4 * half + 8 * (i >> 2)) * LDBP + n];
        }
        // --- A fragments (16x32) + 8 WMMAs ---
#pragma unroll
        for (int tm = 0; tm < 4; ++tm) {
            FragAB af;
            const int m = wm * 64 + tm * 16 + lm;
#pragma unroll
            for (int i = 0; i < 4; ++i) {
                const int k = 2 * i + 8 * half;
                af.u[i]     = *reinterpret_cast<const unsigned int*>(&sA[m * LDA + k]);
                af.u[4 + i] = *reinterpret_cast<const unsigned int*>(&sA[m * LDA + k + 16]);
            }
            acc[tm][0] = wmma_bf16(af, bfr[0], acc[tm][0]);
            acc[tm][1] = wmma_bf16(af, bfr[1], acc[tm][1]);
        }
        __syncthreads();
    }

    // --- fused epilogue ---
#pragma unroll
    for (int tm = 0; tm < 4; ++tm) {
#pragma unroll
        for (int tn = 0; tn < 2; ++tn) {
            const int n = col0 + wn * 32 + tn * 16 + lm;
            float bs = 0.0f;
            if constexpr (BIAS) bs = bias[n];
#pragma unroll
            for (int r = 0; r < 8; ++r) {
                const int m = row0 + wm * 64 + tm * 16 + r + 8 * half;
                float v = acc[tm][tn][r] + bs;
                if constexpr (GELU_ACT) {
                    const float c0 = 0.7978845608028654f;
                    float t = c0 * (v + 0.044715f * v * v * v);
                    v = 0.5f * v * (1.0f + tanhf(t));
                }
                if constexpr (RESID) v += resid[(size_t)m * N + n];
                if constexpr (OUT_BF16)
                    reinterpret_cast<unsigned short*>(outp)[(size_t)m * N + n] = f2bf(v);
                else
                    reinterpret_cast<float*>(outp)[(size_t)m * N + n] = v;
            }
        }
    }
}

// ---------------------------------------------------------------------------
// Flash attention: one wave per (batch, head, 16-query tile).
// S = Q K^T via WMMA (K fragments load straight from global as b128),
// online softmax through per-wave LDS tile (lanes 0..15 each own a query
// row), V staged per wave into LDS packed row-pairs (b128 loads + v_perm),
// O += P V via WMMA with running-max rescale.
// Q/K/V/O are bf16 [ROWS, EMB] with columns = head*64 + d.
// ---------------------------------------------------------------------------
__global__ __launch_bounds__(128) void attention_kernel(
    const unsigned short* __restrict__ Q,
    const unsigned short* __restrict__ Kp,
    const unsigned short* __restrict__ V,
    unsigned short* __restrict__ O)
{
    constexpr int LDV = HDIM + 4;               // dwords per packed V row-pair
    __shared__ float          sS[4][16 * 32];
    __shared__ unsigned short sP[4][16 * 32];
    __shared__ unsigned int   sV[4][16 * LDV];
    __shared__ float          sCorr[4][16];
    __shared__ float          sSum[4][16];

    const int wv   = threadIdx.x >> 5;
    const int lane = threadIdx.x & 31;
    const int half = lane >> 4;
    const int lm   = lane & 15;

    const int gw = blockIdx.x * 4 + wv;         // 0..4095
    const int qt = gw & 127;                    // query tile within sequence
    const int h  = (gw >> 7) & 15;              // head
    const int b  = gw >> 11;                    // batch

    float*          S  = sS[wv];
    unsigned short* P  = sP[wv];
    unsigned int*   Vw = sV[wv];
    float*          CR = sCorr[wv];
    float*          SM = sSum[wv];

    const size_t qrow0 = (size_t)b * SEQ + (size_t)qt * 16;
    const int    col0  = h * HDIM;

    // Q fragments (A layout 16x32, two head-dim halves), loaded once
    FragAB qf[2];
#pragma unroll
    for (int kk = 0; kk < 2; ++kk) {
        const size_t base = (qrow0 + lm) * EMB + col0 + kk * 32;
#pragma unroll
        for (int i = 0; i < 4; ++i) {
            const int k = 2 * i + 8 * half;
            qf[kk].u[i]     = *reinterpret_cast<const unsigned int*>(&Q[base + k]);
            qf[kk].u[4 + i] = *reinterpret_cast<const unsigned int*>(&Q[base + k + 16]);
        }
    }

    f32x8 oacc[4];
#pragma unroll
    for (int g = 0; g < 4; ++g)
#pragma unroll
        for (int r = 0; r < 8; ++r)
            oacc[g][r] = 0.0f;

    float run_max = -3.0e38f;
    float run_sum = 0.0f;

    const int vp = lane >> 1;                   // 0..15 V row pair
    const int vc = (lane & 1) * 32;             // V col half

    const int nj = (qt * 16 + 47) >> 5;         // number of 32-key tiles (causal)

    for (int j = 0; j < nj; ++j) {
        // ---- stage V tile (32 keys x 64 dims) packed as row-pairs ----
        {
            const size_t r0off = ((size_t)b * SEQ + j * 32 + 2 * vp) * EMB + col0 + vc;
            const uint4* g0 = reinterpret_cast<const uint4*>(&V[r0off]);
            const uint4* g1 = reinterpret_cast<const uint4*>(&V[r0off + EMB]);
#pragma unroll
            for (int q4 = 0; q4 < 4; ++q4) {
                uint4 a = g0[q4], c4 = g1[q4];
                uint4 o0, o1;
                o0.x = __builtin_amdgcn_perm(c4.x, a.x, PERM_EVEN);
                o0.y = __builtin_amdgcn_perm(c4.x, a.x, PERM_ODD);
                o0.z = __builtin_amdgcn_perm(c4.y, a.y, PERM_EVEN);
                o0.w = __builtin_amdgcn_perm(c4.y, a.y, PERM_ODD);
                o1.x = __builtin_amdgcn_perm(c4.z, a.z, PERM_EVEN);
                o1.y = __builtin_amdgcn_perm(c4.z, a.z, PERM_ODD);
                o1.z = __builtin_amdgcn_perm(c4.w, a.w, PERM_EVEN);
                o1.w = __builtin_amdgcn_perm(c4.w, a.w, PERM_ODD);
                uint4* s = reinterpret_cast<uint4*>(&Vw[vp * LDV + vc + q4 * 8]);
                s[0] = o0; s[1] = o1;
            }
        }

        // ---- S = Q K^T  (two 16-key column tiles, K-dim = 64 head dims) ----
        f32x8 sacc[2];
#pragma unroll
        for (int kt = 0; kt < 2; ++kt)
#pragma unroll
            for (int r = 0; r < 8; ++r)
                sacc[kt][r] = 0.0f;

#pragma unroll
        for (int kt = 0; kt < 2; ++kt) {
            const size_t krow = ((size_t)b * SEQ + j * 32 + kt * 16 + lm) * EMB + col0;
#pragma unroll
            for (int kk = 0; kk < 2; ++kk) {
                FragAB kb;
#pragma unroll
                for (int i = 0; i < 8; ++i) {
                    const int k = kk * 32 + 8 * half + 2 * (i & 3) + 16 * (i >> 2);
                    kb.u[i] = *reinterpret_cast<const unsigned int*>(&Kp[krow + k]);
                }
                sacc[kt] = wmma_bf16(qf[kk], kb, sacc[kt]);
            }
        }

        // ---- stash scaled scores in LDS ----
#pragma unroll
        for (int kt = 0; kt < 2; ++kt)
#pragma unroll
            for (int r = 0; r < 8; ++r)
                S[(r + 8 * half) * 32 + kt * 16 + lm] = sacc[kt][r] * 0.125f;

        // ---- online softmax: lane m (<16) owns query row m ----
        if (lane < 16) {
            const int m  = lane;
            const int qg = qt * 16 + m;
            float mx = -3.0e38f;
            float sv[32];
#pragma unroll
            for (int n = 0; n < 32; ++n) {
                float xv = S[m * 32 + n];
                const int kg = j * 32 + n;
                xv = (kg <= qg) ? xv : -3.0e38f;
                sv[n] = xv;
                mx = fmaxf(mx, xv);
            }
            const float nm   = fmaxf(run_max, mx);
            const float corr = __expf(run_max - nm);
            float s = 0.0f;
#pragma unroll
            for (int n = 0; n < 32; ++n) {
                const float e = __expf(sv[n] - nm);
                s += e;
                P[m * 32 + n] = f2bf(e);
            }
            run_sum = run_sum * corr + s;
            run_max = nm;
            CR[m] = corr;
        }

        // ---- rescale O accumulators by per-row correction ----
        float cr[8];
#pragma unroll
        for (int r = 0; r < 8; ++r) cr[r] = CR[r + 8 * half];
#pragma unroll
        for (int g = 0; g < 4; ++g)
#pragma unroll
            for (int r = 0; r < 8; ++r)
                oacc[g][r] *= cr[r];

        // ---- P fragment (A layout 16x32: 16 queries x 32 keys) ----
        FragAB pf;
#pragma unroll
        for (int i = 0; i < 4; ++i) {
            const int k = 2 * i + 8 * half;
            pf.u[i]     = *reinterpret_cast<const unsigned int*>(&P[lm * 32 + k]);
            pf.u[4 + i] = *reinterpret_cast<const unsigned int*>(&P[lm * 32 + k + 16]);
        }

        // ---- O += P V  (4 d-column groups of 16, packed V in LDS) ----
#pragma unroll
        for (int g = 0; g < 4; ++g) {
            FragAB vb;
#pragma unroll
            for (int i = 0; i < 8; ++i)
                vb.u[i] = Vw[((i & 3) + 4 * half + 8 * (i >> 2)) * LDV + g * 16 + lm];
            oacc[g] = wmma_bf16(pf, vb, oacc[g]);
        }
    }

    if (lane < 16) SM[lane] = run_sum;

    float inv[8];
#pragma unroll
    for (int r = 0; r < 8; ++r) inv[r] = 1.0f / SM[r + 8 * half];

#pragma unroll
    for (int g = 0; g < 4; ++g)
#pragma unroll
        for (int r = 0; r < 8; ++r) {
            const size_t row = qrow0 + r + 8 * half;
            O[row * EMB + col0 + g * 16 + lm] = f2bf(oacc[g][r] * inv[r]);
        }
}

// ---------------------------------------------------------------------------
// Launcher
// ---------------------------------------------------------------------------
extern "C" void kernel_launch(void* const* d_in, const int* in_sizes, int n_in,
                              void* d_out, int out_size, void* d_ws, size_t ws_size,
                              hipStream_t stream)
{
    (void)in_sizes; (void)n_in; (void)out_size; (void)ws_size;

    const float* x  = (const float*)d_in[0];
    const float* g1 = (const float*)d_in[1];
    const float* s1 = (const float*)d_in[2];
    const float* Wq = (const float*)d_in[3];
    const float* Wk = (const float*)d_in[4];
    const float* Wv = (const float*)d_in[5];
    const float* Wo = (const float*)d_in[6];
    const float* bo = (const float*)d_in[7];
    const float* g2 = (const float*)d_in[8];
    const float* s2 = (const float*)d_in[9];
    const float* W1 = (const float*)d_in[10];
    const float* b1 = (const float*)d_in[11];
    const float* W2 = (const float*)d_in[12];
    const float* b2 = (const float*)d_in[13];
    float* out = (float*)d_out;

    char* w = (char*)d_ws;
    auto take = [&](size_t bytes) -> char* {
        char* p = w;
        w += (bytes + 255) & ~(size_t)255;
        return p;
    };

    typedef unsigned short u16;
    u16* wqB  = (u16*)take((size_t)EMB * EMB * 2);
    u16* wkB  = (u16*)take((size_t)EMB * EMB * 2);
    u16* wvB  = (u16*)take((size_t)EMB * EMB * 2);
    u16* woB  = (u16*)take((size_t)EMB * EMB * 2);
    u16* w1B  = (u16*)take((size_t)EMB * FFDIM * 2);
    u16* w2B  = (u16*)take((size_t)FFDIM * EMB * 2);
    u16* h1   = (u16*)take((size_t)ROWS * EMB * 2);
    u16* qB   = (u16*)take((size_t)ROWS * EMB * 2);
    u16* kB   = (u16*)take((size_t)ROWS * EMB * 2);
    u16* vB   = (u16*)take((size_t)ROWS * EMB * 2);
    u16* ctxB = (u16*)take((size_t)ROWS * EMB * 2);
    float* x1 = (float*)take((size_t)ROWS * EMB * 4);
    u16* h2   = (u16*)take((size_t)ROWS * EMB * 2);
    u16* midB = (u16*)take((size_t)ROWS * FFDIM * 2);

    const dim3 blk256(256);
    const int nSq = EMB * EMB / 4;     // 262144
    const int nFf = EMB * FFDIM / 4;   // 1048576

    // 1) weights -> bf16
    convert_bf16_kernel<<<dim3((nSq + 255) / 256), blk256, 0, stream>>>(Wq, wqB, nSq);
    convert_bf16_kernel<<<dim3((nSq + 255) / 256), blk256, 0, stream>>>(Wk, wkB, nSq);
    convert_bf16_kernel<<<dim3((nSq + 255) / 256), blk256, 0, stream>>>(Wv, wvB, nSq);
    convert_bf16_kernel<<<dim3((nSq + 255) / 256), blk256, 0, stream>>>(Wo, woB, nSq);
    convert_bf16_kernel<<<dim3((nFf + 255) / 256), blk256, 0, stream>>>(W1, w1B, nFf);
    convert_bf16_kernel<<<dim3((nFf + 255) / 256), blk256, 0, stream>>>(W2, w2B, nFf);

    // 2) LN1
    layernorm_bf16_kernel<<<dim3(ROWS), blk256, 0, stream>>>(x, g1, s1, h1);

    // 3) QKV projections (bf16 out)
    const dim3 gE(EMB / 128, ROWS / 128);      // (8, 32)
    gemm_bf16_kernel<true, false, false, false><<<gE, blk256, 0, stream>>>(h1, wqB, nullptr, nullptr, qB, ROWS, EMB, EMB);
    gemm_bf16_kernel<true, false, false, false><<<gE, blk256, 0, stream>>>(h1, wkB, nullptr, nullptr, kB, ROWS, EMB, EMB);
    gemm_bf16_kernel<true, false, false, false><<<gE, blk256, 0, stream>>>(h1, wvB, nullptr, nullptr, vB, ROWS, EMB, EMB);

    // 4) causal flash attention
    attention_kernel<<<dim3(1024), dim3(128), 0, stream>>>(qB, kB, vB, ctxB);

    // 5) out projection + bias + residual -> x1 (f32)
    gemm_bf16_kernel<false, true, false, true><<<gE, blk256, 0, stream>>>(ctxB, woB, bo, x, x1, ROWS, EMB, EMB);

    // 6) LN2
    layernorm_bf16_kernel<<<dim3(ROWS), blk256, 0, stream>>>(x1, g2, s2, h2);

    // 7) FFN1 + bias + GELU (bf16 out)
    const dim3 gF(FFDIM / 128, ROWS / 128);    // (32, 32)
    gemm_bf16_kernel<true, true, true, false><<<gF, blk256, 0, stream>>>(h2, w1B, b1, nullptr, midB, ROWS, FFDIM, EMB);

    // 8) FFN2 + bias + residual -> out (f32)
    gemm_bf16_kernel<false, true, false, true><<<gE, blk256, 0, stream>>>(midB, w2B, b2, x1, out, ROWS, EMB, FFDIM);
}
